// SampledDenseEdgeConv_89532888252869
// MI455X (gfx1250) — matrix operand order repaired and checked by reading, not compiled, not run on hardware
//
#include <hip/hip_runtime.h>
#include <hip/hip_bf16.h>
#include <stddef.h>

// ---------------------------------------------------------------------------
// SampledDenseEdgeConv for MI455X (gfx1250, wave32, WMMA 16x16x32 f16)
// B=4, C=64, N=8192, NS=2048, K=16, g=64.
// ~16 GFLOP of GEMM vs ~10MB live data -> L2-resident; never materialize the
// 256MB distance matrix. Feature arrays are stored POINT-MAJOR (f16) so every
// WMMA fragment is a contiguous aligned 16/32B run (b128 loads, no b16 packs).
// ---------------------------------------------------------------------------

#define NB   4
#define NC   64
#define NP   8192
#define NS   2048
#define KNN  17      // K+1, self dropped at the end
#define KOUT 16

typedef __attribute__((ext_vector_type(16))) _Float16 v16h;
typedef __attribute__((ext_vector_type(8)))  _Float16 v8h;
typedef __attribute__((ext_vector_type(8)))  float    v8f;
typedef __attribute__((ext_vector_type(4)))  int      v4i;

#if defined(__has_builtin)
#  if __has_builtin(__builtin_amdgcn_global_load_async_to_lds_b128) && \
      __has_builtin(__builtin_amdgcn_s_wait_asynccnt)
#    define USE_ASYNC_LDS 1
#  endif
#endif
#ifndef USE_ASYNC_LDS
#  define USE_ASYNC_LDS 0
#endif

#if USE_ASYNC_LDS
typedef __attribute__((address_space(1))) v4i* gas_b128_t;   // global int4*
typedef __attribute__((address_space(3))) v4i* las_b128_t;   // LDS int4*
#endif

__device__ __forceinline__ v16h cat8(v8h lo, v8h hi) {
  return __builtin_shufflevector(lo, hi, 0, 1, 2, 3, 4, 5, 6, 7,
                                 8, 9, 10, 11, 12, 13, 14, 15);
}

// ---------------------------------------------------------------------------
// Kernel 1: transposing convert: xht[b][n][c] = (f16)x[b][c][n], pp = |p|^2,
// plus f16 copies of W0/W1/W2 (row-major o,i — A-fragment friendly).
// ---------------------------------------------------------------------------
__global__ void prep_convert_kernel(const float* __restrict__ x,
                                    const float* __restrict__ W0,
                                    const float* __restrict__ W1,
                                    const float* __restrict__ W2,
                                    _Float16* __restrict__ xht,
                                    _Float16* __restrict__ w0h,
                                    _Float16* __restrict__ w1h,
                                    _Float16* __restrict__ w2h,
                                    float* __restrict__ pp) {
  int tid = blockIdx.x * blockDim.x + threadIdx.x;
  int stride = gridDim.x * blockDim.x;
  for (int t = tid; t < NB * NP; t += stride) {     // t = b*NP + n
    int b = t >> 13, n = t & (NP - 1);
    const float* xb = x + (size_t)b * NC * NP + n;
    _Float16* dst = xht + (size_t)t * NC;
    float s = 0.f;
    for (int c = 0; c < NC; ++c) {
      float v = xb[(size_t)c * NP];
      s += v * v;
      dst[c] = (_Float16)v;
    }
    pp[t] = s;
  }
  for (int t = tid; t < 64 * 128; t += stride) w0h[t] = (_Float16)W0[t];
  for (int t = tid; t < 64 * 128; t += stride) w1h[t] = (_Float16)W1[t];
  for (int t = tid; t < 64 * 192; t += stride) w2h[t] = (_Float16)W2[t];
}

// ---------------------------------------------------------------------------
// Kernel 2: farthest point sampling. One workgroup (1024 thr = 32 waves) per
// batch; coords + running min-dist live in registers; 2047 dependent
// iterations of a two-level wave32 argmax reduction. Ties -> lowest index
// (matches jnp.argmax).
// ---------------------------------------------------------------------------
__global__ __launch_bounds__(1024) void fps_kernel(const float* __restrict__ xyz,
                                                   int* __restrict__ sidx) {
  int b = blockIdx.x;
  const float* px = xyz + (size_t)b * 3 * NP;
  const float* py = px + NP;
  const float* pz = py + NP;
  int tid = threadIdx.x;

  float cx[8], cy[8], cz[8], dist[8];
#pragma unroll
  for (int j = 0; j < 8; ++j) {
    int n = tid + j * 1024;
    cx[j] = px[n]; cy[j] = py[n]; cz[j] = pz[n];
    dist[j] = 3.402823466e+38f;
  }

  __shared__ float wval[32];
  __shared__ int   widx[32];
  __shared__ int   curLast;
  if (tid == 0) { sidx[b * NS] = 0; curLast = 0; }
  __syncthreads();

  int last = 0;
  int lane = tid & 31;
  int wv   = tid >> 5;
  for (int s = 1; s < NS; ++s) {
    float lx = px[last], ly = py[last], lz = pz[last];  // L2-resident broadcast
    float v = -1.0f; int vi = 0x7fffffff;
#pragma unroll
    for (int j = 0; j < 8; ++j) {
      float dx = cx[j] - lx, dy = cy[j] - ly, dz = cz[j] - lz;
      float d = dx * dx + dy * dy + dz * dz;
      d = fminf(dist[j], d);
      dist[j] = d;
      int n = tid + j * 1024;
      if (d > v || (d == v && n < vi)) { v = d; vi = n; }
    }
#pragma unroll
    for (int off = 16; off > 0; off >>= 1) {
      float ov = __shfl_down(v, off, 32);
      int   oi = __shfl_down(vi, off, 32);
      if (ov > v || (ov == v && oi < vi)) { v = ov; vi = oi; }
    }
    if (lane == 0) { wval[wv] = v; widx[wv] = vi; }
    __syncthreads();
    if (wv == 0) {
      v = wval[lane]; vi = widx[lane];
#pragma unroll
      for (int off = 16; off > 0; off >>= 1) {
        float ov = __shfl_down(v, off, 32);
        int   oi = __shfl_down(vi, off, 32);
        if (ov > v || (ov == v && oi < vi)) { v = ov; vi = oi; }
      }
      if (lane == 0) { curLast = vi; sidx[b * NS + s] = vi; }
    }
    __syncthreads();
    last = curLast;
  }
}

// ---------------------------------------------------------------------------
// Kernel 3: gather sampled features (point-major f16 rows) + qq = |q|^2, and
// emit the sampled_xyz / sampled_idx(as float) output sections.
// ---------------------------------------------------------------------------
__global__ void prep_gather_kernel(const float* __restrict__ xyz,
                                   const int* __restrict__ sidx,
                                   const _Float16* __restrict__ xht,
                                   const float* __restrict__ pp,
                                   _Float16* __restrict__ qht,
                                   float* __restrict__ qq,
                                   float* __restrict__ out_sxyz,
                                   float* __restrict__ out_sidx) {
  int t = blockIdx.x * blockDim.x + threadIdx.x;
  if (t >= NB * NS) return;
  int b = t / NS, m = t % NS;
  int idx = sidx[t];
  qq[t] = pp[b * NP + idx];
  const v8h* src = (const v8h*)(xht + ((size_t)b * NP + idx) * NC);
  v8h* dst = (v8h*)(qht + (size_t)t * NC);
  for (int c = 0; c < NC / 8; ++c) dst[c] = src[c];
  for (int d = 0; d < 3; ++d)
    out_sxyz[((size_t)b * 3 + d) * NS + m] = xyz[((size_t)b * 3 + d) * NP + idx];
  out_sidx[t] = (float)idx;
}

// ---------------------------------------------------------------------------
// Kernel 4: fused KNN. Workgroup = 16 waves = one 16-query tile.
//  Stage A: S = Q(16x64) * P_chunk(64x256) via v_wmma_f32_16x16x32_f16,
//           one 16x16 column tile per wave; the 32KB P-chunk is staged into
//           LDS with global_load_async_to_lds_b128 (+ s_wait_asynccnt) when
//           available. d = qq + pp - 2S into LDS.
//  Stage B: wave w keeps query-row w's running top-17 with a threshold-
//           pruned tournament (chunk-min vs current 17th best).
// ---------------------------------------------------------------------------
__global__ __launch_bounds__(512) void knn_kernel(const _Float16* __restrict__ qht,
                                                  const _Float16* __restrict__ xht,
                                                  const float* __restrict__ qq,
                                                  const float* __restrict__ pp,
                                                  int* __restrict__ knn_idx) {
  int b    = blockIdx.x >> 7;           // 128 query tiles per batch
  int m0   = (blockIdx.x & 127) * 16;
  int lane = threadIdx.x & 31;
  int wv   = threadIdx.x >> 5;          // 0..15

  __shared__ float sd[16 * 256];
  __shared__ float topv[16][KNN];
  __shared__ int   topi[16][KNN];
#if USE_ASYNC_LDS
  __shared__ __align__(32) _Float16 stage[256 * NC];   // one P chunk, 32KB
#endif

  // ---- A fragments (16x32 f16): lane<16 holds K 0-15, lane>=16 K 16-31
  int am  = lane & 15;
  int ahi = (lane >> 4) * 16;
  const _Float16* qrow = qht + ((size_t)b * NS + m0 + am) * NC;
  v16h a0 = cat8(*(const v8h*)(qrow + ahi),      *(const v8h*)(qrow + 8 + ahi));
  v16h a1 = cat8(*(const v8h*)(qrow + 32 + ahi), *(const v8h*)(qrow + 40 + ahi));

  // C/D layout: lane holds column cn = lane&15, rows mb..mb+7
  int cn = lane & 15;
  int mb = (lane >> 4) * 8;
  float qqr[8];
#pragma unroll
  for (int r = 0; r < 8; ++r) qqr[r] = qq[b * NS + m0 + mb + r];

  const _Float16* pbase = xht + (size_t)b * NP * NC;   // point-major rows
  int bk = (lane >> 4) * 16;            // B frag: lane<16 K 0-15, lane>=16 K 16-31

  int   count  = 0;
  float worstV = -3.402823466e+38f;
  int   worstI = -1, worstP = 0;

  for (int ch = 0; ch < NP / 256; ++ch) {
    int n0 = ch * 256;
    int nt = n0 + wv * 16;

#if USE_ASYNC_LDS
    {
      // 256 rows * 64 halves = 2048 x 16B units; 4 units per thread
      const _Float16* gsrc = pbase + (size_t)n0 * NC;
#pragma unroll
      for (int j = 0; j < 4; ++j) {
        int u = threadIdx.x * 4 + j;
        __builtin_amdgcn_global_load_async_to_lds_b128(
            (gas_b128_t)(gsrc + u * 8),
            (las_b128_t)(stage + u * 8), 0, 0);
      }
      __builtin_amdgcn_s_wait_asynccnt(0);
    }
    __syncthreads();
    const _Float16* brow = stage + (wv * 16 + cn) * NC;
#else
    const _Float16* brow = pbase + (size_t)(nt + cn) * NC;
    if (ch + 1 < NP / 256)              // global_prefetch_b8 next chunk
      __builtin_prefetch(brow + 256 * NC, 0, 0);
#endif
    v16h b0 = *(const v16h*)(brow + bk);        // K 0..31
    v16h b1 = *(const v16h*)(brow + 32 + bk);   // K 32..63

    v8f acc = {};
    acc = __builtin_amdgcn_wmma_f32_16x16x32_f16(false, a0, false, b0,
                                                 (short)0, acc, false, false);
    acc = __builtin_amdgcn_wmma_f32_16x16x32_f16(false, a1, false, b1,
                                                 (short)0, acc, false, false);

    float ppn = pp[b * NP + nt + cn];
#pragma unroll
    for (int r = 0; r < 8; ++r)
      sd[(mb + r) * 256 + wv * 16 + cn] = qqr[r] + ppn - 2.0f * acc[r];
    __syncthreads();

    // ---- Stage B: wave wv selects for query row wv over this 256 chunk
    float dloc[8]; int gidx[8];
#pragma unroll
    for (int j = 0; j < 8; ++j) {
      int nl = lane + j * 32;
      dloc[j] = sd[wv * 256 + nl];
      gidx[j] = n0 + nl;
    }
    unsigned cons = 0;
    for (;;) {
      float v = 3.402823466e+38f; int gi = 0x7fffffff; int slot = -1;
#pragma unroll
      for (int j = 0; j < 8; ++j) {
        if (!(cons & (1u << j)) &&
            (dloc[j] < v || (dloc[j] == v && gidx[j] < gi))) {
          v = dloc[j]; gi = gidx[j]; slot = j;
        }
      }
      int ln = lane;
#pragma unroll
      for (int off = 16; off > 0; off >>= 1) {
        float ov  = __shfl_down(v, off, 32);
        int   ogi = __shfl_down(gi, off, 32);
        int   oln = __shfl_down(ln, off, 32);
        int   osl = __shfl_down(slot, off, 32);
        if (ov < v || (ov == v && ogi < gi)) { v = ov; gi = ogi; ln = oln; slot = osl; }
      }
      int accept = 0;
      if (lane == 0 && slot >= 0) {
        if (count < KNN) {
          topv[wv][count] = v; topi[wv][count] = gi;
          if (v > worstV || (v == worstV && gi > worstI)) { worstV = v; worstI = gi; worstP = count; }
          count++; accept = 1;
        } else if (v < worstV || (v == worstV && gi < worstI)) {
          topv[wv][worstP] = v; topi[wv][worstP] = gi;
          worstV = -3.402823466e+38f; worstI = -1;
          for (int q = 0; q < KNN; ++q) {
            float tv = topv[wv][q]; int ti = topi[wv][q];
            if (tv > worstV || (tv == worstV && ti > worstI)) { worstV = tv; worstI = ti; worstP = q; }
          }
          accept = 1;
        }
      }
      accept = __shfl(accept, 0, 32);
      if (!accept) break;                 // chunk-min >= 17th best -> prune
      int wl  = __shfl(ln, 0, 32);
      int wsl = __shfl(slot, 0, 32);
      if (lane == wl && wsl >= 0) cons |= (1u << wsl);
    }
    __syncthreads();
  }

  // sort 17 ascending (val, idx), drop self (entry 0), emit 16 neighbors
  if (lane == 0) {
    for (int i = 1; i < KNN; ++i) {
      float v = topv[wv][i]; int gi = topi[wv][i];
      int j = i - 1;
      while (j >= 0 && (topv[wv][j] > v || (topv[wv][j] == v && topi[wv][j] > gi))) {
        topv[wv][j + 1] = topv[wv][j]; topi[wv][j + 1] = topi[wv][j]; --j;
      }
      topv[wv][j + 1] = v; topi[wv][j + 1] = gi;
    }
    for (int k = 0; k < KOUT; ++k)
      knn_idx[(b * NS + m0 + wv) * KOUT + k] = topi[wv][k + 1];
  }
}

// ---------------------------------------------------------------------------
// Kernel 5: fused EdgeConv stack. 4 sampled points per workgroup (64 cols).
// Activation matrix Yt is [col][chan] (64 x 256 f16, 32KB LDS) so every B
// fragment is ONE aligned v16h LDS load and every output tile ONE v8h store.
// Channel windows: [192,256)=center, [128,192)=diff->relu0, [64,128)=relu1,
// [0,64)=conv2. K-runs never cross the 64-boundary, so each (k0,bk) run maps
// to one contiguous window. Layer/K loops fully unrolled -> 28 static WMMAs.
// ---------------------------------------------------------------------------
__global__ __launch_bounds__(256) void edgeconv_kernel(
    const _Float16* __restrict__ qht, const _Float16* __restrict__ xht,
    const int* __restrict__ knn_idx,
    const _Float16* __restrict__ w0h, const _Float16* __restrict__ w1h,
    const _Float16* __restrict__ w2h,
    const float* __restrict__ b0, const float* __restrict__ b1,
    const float* __restrict__ b2, float* __restrict__ yout) {
  int b  = blockIdx.x >> 9;             // 512 groups per batch
  int m0 = (blockIdx.x & 511) * 4;
  int tid  = threadIdx.x;
  int lane = tid & 31;
  int wv   = tid >> 5;                  // 0..7

  __shared__ __align__(32) _Float16 Yt[64 * 256];
  __shared__ int nb[64];

  if (tid < 64) nb[tid] = knn_idx[(b * NS + m0 + (tid >> 4)) * KOUT + (tid & 15)];
  __syncthreads();

  const _Float16* qb = qht + (size_t)b * NS * NC;
  const _Float16* xb = xht + (size_t)b * NP * NC;
  for (int t = tid; t < 64 * 64; t += 256) {
    int col = t >> 6, c = t & 63, p = col >> 4;
    _Float16 ctr = qb[((size_t)m0 + p) * NC + c];
    _Float16 nbv = xb[(size_t)nb[col] * NC + c];
    Yt[col * 256 + 192 + c] = ctr;
    Yt[col * 256 + 128 + c] = (_Float16)((float)nbv - (float)ctr);
  }
  __syncthreads();

  int am  = lane & 15, ahi = (lane >> 4) * 16;
  int cn  = lane & 15, bk  = (lane >> 4) * 16;
  int mbs = (lane >> 4) * 8;

#pragma unroll
  for (int layer = 0; layer < 3; ++layer) {
    const _Float16* W = (layer == 0) ? w0h : (layer == 1) ? w1h : w2h;
    const float* bias = (layer == 0) ? b0  : (layer == 1) ? b1  : b2;
    const int Ki      = (layer == 2) ? 192 : 128;
    const int outBase = (layer == 0) ? 128 : (layer == 1) ? 64 : 0;

    v8f acc[2] = {{}, {}};
#pragma unroll
    for (int t = 0; t < 2; ++t) {
      int tile = wv * 2 + t;
      int mt = tile >> 2, ntile = tile & 3;       // 4 M-tiles x 4 N-tiles
      const _Float16* wrow = W + (size_t)(mt * 16 + am) * Ki;
      const _Float16* ycol = Yt + (ntile * 16 + cn) * 256;
#pragma unroll
      for (int k0 = 0; k0 < Ki; k0 += 32) {
        v16h A = cat8(*(const v8h*)(wrow + k0 + ahi),
                      *(const v8h*)(wrow + k0 + 8 + ahi));
        int i0  = k0 + bk;                        // run start; never crosses 64
        int row = (layer == 0) ? ((i0 < 64) ? 192 + i0 : 64 + i0)
                               : ((layer == 1) ? 128 + i0 : 64 + i0);
        v16h Bf = *(const v16h*)(ycol + row);
        acc[t] = __builtin_amdgcn_wmma_f32_16x16x32_f16(false, A, false, Bf,
                                                        (short)0, acc[t], false, false);
      }
    }
    __syncthreads();                     // all reads of input windows done
#pragma unroll
    for (int t = 0; t < 2; ++t) {
      int tile = wv * 2 + t;
      int mt = tile >> 2, ntile = tile & 3;
      v8h outv;
#pragma unroll
      for (int r = 0; r < 8; ++r) {
        int o = mt * 16 + mbs + r;
        float v = acc[t][r] + bias[o];
        if (layer < 2) v = fmaxf(v, 0.0f);
        outv[r] = (_Float16)v;
      }
      *(v8h*)(Yt + (ntile * 16 + cn) * 256 + outBase + mt * 16 + mbs) = outv;
    }
    __syncthreads();
  }

  // max over k (16 cols per point); channel order [conv2|relu1|relu0|center]
  int chn = tid;                         // 0..255
  for (int p = 0; p < 4; ++p) {
    float mx = -3.402823466e+38f;
    for (int k = 0; k < KOUT; ++k)
      mx = fmaxf(mx, (float)Yt[(p * 16 + k) * 256 + chn]);
    yout[((size_t)b * 256 + chn) * NS + m0 + p] = mx;
  }
}

// ---------------------------------------------------------------------------
// Launch
// ---------------------------------------------------------------------------
extern "C" void kernel_launch(void* const* d_in, const int* in_sizes, int n_in,
                              void* d_out, int out_size, void* d_ws, size_t ws_size,
                              hipStream_t stream) {
  (void)in_sizes; (void)n_in; (void)out_size; (void)ws_size;
  const float* x   = (const float*)d_in[0];
  const float* xyz = (const float*)d_in[1];
  const float* W0  = (const float*)d_in[2];
  const float* b0  = (const float*)d_in[3];
  const float* W1  = (const float*)d_in[4];
  const float* b1  = (const float*)d_in[5];
  const float* W2  = (const float*)d_in[6];
  const float* b2  = (const float*)d_in[7];
  float* out = (float*)d_out;

  // workspace layout (~5.8 MB)
  char* ws = (char*)d_ws;
  int*      sidx = (int*)ws;                               // 4*2048
  float*    qq   = (float*)(ws + 32768);                   // 4*2048
  float*    pp   = (float*)(ws + 65536);                   // 4*8192
  _Float16* xht  = (_Float16*)(ws + 196608);               // 4*8192*64 (point-major)
  _Float16* qht  = (_Float16*)(ws + 4390912);              // 4*2048*64 (point-major)
  int*      knn  = (int*)(ws + 5439488);                   // 4*2048*16
  _Float16* w0h  = (_Float16*)(ws + 5963776);              // 64*128
  _Float16* w1h  = (_Float16*)(ws + 5980160);              // 64*128
  _Float16* w2h  = (_Float16*)(ws + 5996544);              // 64*192

  const size_t Y_ELEMS = (size_t)NB * 256 * NS;            // 2,097,152
  float* out_sxyz = out + Y_ELEMS;                         // 4*3*2048
  float* out_sidx = out + Y_ELEMS + (size_t)NB * 3 * NS;   // 4*2048 (as float)

  prep_convert_kernel<<<2048, 256, 0, stream>>>(x, W0, W1, W2,
                                                xht, w0h, w1h, w2h, pp);
  fps_kernel<<<NB, 1024, 0, stream>>>(xyz, sidx);
  prep_gather_kernel<<<(NB * NS + 255) / 256, 256, 0, stream>>>(
      xyz, sidx, xht, pp, qht, qq, out_sxyz, out_sidx);
  knn_kernel<<<NB * (NS / 16), 512, 0, stream>>>(qht, xht, qq, pp, knn);
  edgeconv_kernel<<<NB * (NS / 4), 256, 0, stream>>>(
      qht, xht, knn, w0h, w1h, w2h, b0, b1, b2, out);
}
